// DGLJTNNEncoder_75771813036634
// MI455X (gfx1250) — matrix operand order repaired and checked by reading, not compiled, not run on hardware
//
#include <hip/hip_runtime.h>
#include <hip/hip_bf16.h>
#include <stdint.h>

// ---------------- problem constants (static per reference) ----------------
#define Bn   2048
#define Hn   450
#define NTn  20
#define Nn   (Bn * NTn)            // 40960 nodes
#define En   (2 * Bn * (NTn - 1))  // 77824 edges (+1 dummy pad)
#define PMAX (Bn * 5)              // 10240 max edges per level
#define NLEV 10
#define KP2  928                   // 2H=900 padded to mult of 32
#define KP1  480                   // H=450 padded to mult of 32
#define NPAD 480                   // H padded: 30 tiles of 16
#define NCT  (NPAD / 16)           // 30 column tiles
#define NTW  6                     // column tiles per wave (A reuse factor)
#define KT2  (KP2 / 32)            // 29 k-tiles for 2H panels
#define KT1  (KP1 / 32)            // 15 k-tiles for H panels

typedef __attribute__((ext_vector_type(16))) _Float16 v16h;
typedef __attribute__((ext_vector_type(8)))  _Float16 v8h;
typedef __attribute__((ext_vector_type(8)))  float    v8f;

// ---------------- WMMA fragment loaders (ISA 7.12.2 layouts, wave32) ------
// A: 16x32 f16 from row-major source, leading dim ld (halfs). Per lane this is
// two contiguous 16B runs -> global_load_b128 x2.
__device__ __forceinline__ v16h wmma_load_a(const _Float16* __restrict__ A, int ld) {
    int lane = threadIdx.x & 31;
    int m = lane & 15, hi = lane >> 4;
    const _Float16* row = A + (size_t)m * ld + hi * 8;
    v8h lo = *(const v8h*)(row);
    v8h hiv = *(const v8h*)(row + 16);
    v16h a;
#pragma unroll
    for (int j = 0; j < 8; ++j) { a[j] = lo[j]; a[8 + j] = hiv[j]; }
    return a;
}
// B: fragment-major weight panel. Each lane's 16 halfs stored contiguously:
// F[(((kt*NCT)+ct)*32 + lane)*16 + j]  with K = kt*32 + (lane>>4)*8 + (j&7) + (j>>3)*16,
// N = ct*16 + (lane&15). One aligned 32B per-lane load (b128 x2, coalesced).
__device__ __forceinline__ v16h wmma_load_bfrag(const _Float16* __restrict__ F,
                                                int kt, int ct) {
    int lane = threadIdx.x & 31;
    return *(const v16h*)(F + (((size_t)kt * NCT + ct) * 32 + lane) * 16);
}

__device__ __forceinline__ float fast_sigmoid(float x) {
    return __fdividef(1.0f, 1.0f + __expf(-x));
}
__device__ __forceinline__ float fast_tanh(float x) {
    float e2 = __expf(2.0f * x);                 // inf -> 1, 0 -> -1 : saturates
    return 1.0f - __fdividef(2.0f, e2 + 1.0f);
}

// ---------------- kernels -------------------------------------------------
__global__ __launch_bounds__(256)
void embed_kernel(const int* __restrict__ wid, const float* __restrict__ emb,
                  float* __restrict__ xp) {
    size_t idx = (size_t)blockIdx.x * blockDim.x + threadIdx.x;
    if (idx >= (size_t)Nn * Hn) return;
    int n = (int)(idx / Hn), c = (int)(idx % Hn);
    xp[idx] = emb[(size_t)wid[n] * Hn + c];
}

// f32 [K x N] row-major -> f16 fragment-major panel (zero padded)
__global__ __launch_bounds__(256)
void convert_frag_kernel(const float* __restrict__ W, _Float16* __restrict__ F,
                         int K, int N, int KT) {
    size_t idx = (size_t)blockIdx.x * blockDim.x + threadIdx.x;
    size_t total = (size_t)KT * NCT * 512;       // 32 lanes * 16 halfs per (kt,ct)
    if (idx >= total) return;
    int j    = (int)(idx & 15);
    int lane = (int)((idx >> 4) & 31);
    int ct   = (int)((idx >> 9) % NCT);
    int kt   = (int)(idx / ((size_t)NCT * 512));
    int hi = lane >> 4, nl = lane & 15;
    int k = kt * 32 + hi * 8 + (j & 7) + ((j >> 3) << 4);
    int n = ct * 16 + nl;
    float v = (k < K && n < N) ? W[(size_t)k * N + n] : 0.0f;
    F[idx] = (_Float16)v;
}

// Gather per-edge A panels for one level: cat1=[sx|s], cat2=[sx|arm], dx16, sbuf(f32 s)
__global__ __launch_bounds__(256)
void stage_level_kernel(const int* __restrict__ lev_row, const int* __restrict__ src,
                        const int* __restrict__ dstA, const int* __restrict__ rev,
                        const float* __restrict__ xp,
                        const float* __restrict__ node_m, const float* __restrict__ node_rm,
                        const float* __restrict__ mbuf, const float* __restrict__ rmbuf,
                        _Float16* __restrict__ cat1, _Float16* __restrict__ cat2,
                        _Float16* __restrict__ dx16, float* __restrict__ sbuf) {
    int p   = blockIdx.x;
    int eid = lev_row[p];
    int s_  = src[eid], d_ = dstA[eid], rv = rev[eid];
    const float* sxp = xp + (size_t)s_ * Hn;
    const float* dxp = xp + (size_t)d_ * Hn;
    const float* nm  = node_m  + (size_t)s_ * Hn;
    const float* nrm = node_rm + (size_t)s_ * Hn;
    const float* mr  = mbuf  + (size_t)rv * Hn;
    const float* rmr = rmbuf + (size_t)rv * Hn;
    for (int c = threadIdx.x; c < KP2; c += blockDim.x) {
        _Float16 v1, v2;
        if (c < Hn) {
            float x = sxp[c]; v1 = (_Float16)x; v2 = v1;
        } else if (c < 2 * Hn) {
            int k = c - Hn;
            float s = nm[k] - mr[k];
            float a = nrm[k] - rmr[k];
            sbuf[(size_t)p * Hn + k] = s;
            v1 = (_Float16)s; v2 = (_Float16)a;
        } else {
            v1 = (_Float16)0.0f; v2 = v1;
        }
        cat1[(size_t)p * KP2 + c] = v1;
        cat2[(size_t)p * KP2 + c] = v2;
    }
    for (int c = threadIdx.x; c < KP1; c += blockDim.x)
        dx16[(size_t)p * KP1 + c] = (c < Hn) ? (_Float16)dxp[c] : (_Float16)0.0f;
}

// z/h GEMMs + GRU combine: m_new = (1-z)*s + z*tanh(...). NTW col tiles / wave.
__global__ __launch_bounds__(32)
void level_zh_kernel(const _Float16* __restrict__ cat1, const _Float16* __restrict__ cat2,
                     const _Float16* __restrict__ WzF, const _Float16* __restrict__ WhF,
                     const float* __restrict__ bz, const float* __restrict__ bh,
                     const float* __restrict__ sbuf, const int* __restrict__ lev_row,
                     float* __restrict__ mbuf, _Float16* __restrict__ mnew16) {
    int ctg = blockIdx.x, rt = blockIdx.y;
    int lane = threadIdx.x & 31;
    int n = lane & 15, hi = lane >> 4;
    int row0 = rt * 16;
    v8f accZ[NTW] = {}, accH[NTW] = {};
    const _Float16* a1 = cat1 + (size_t)row0 * KP2;
    const _Float16* a2 = cat2 + (size_t)row0 * KP2;
    for (int kt = 0; kt < KT2; ++kt) {
        v16h A1 = wmma_load_a(a1 + kt * 32, KP2);
        v16h A2 = wmma_load_a(a2 + kt * 32, KP2);
#pragma unroll
        for (int t = 0; t < NTW; ++t) {
            int ct = ctg * NTW + t;
            v16h Bz = wmma_load_bfrag(WzF, kt, ct);
            accZ[t] = __builtin_amdgcn_wmma_f32_16x16x32_f16(false, A1, false, Bz, (short)0, accZ[t], false, false);
            v16h Bh = wmma_load_bfrag(WhF, kt, ct);
            accH[t] = __builtin_amdgcn_wmma_f32_16x16x32_f16(false, A2, false, Bh, (short)0, accH[t], false, false);
        }
    }
#pragma unroll
    for (int t = 0; t < NTW; ++t) {
        int col = (ctg * NTW + t) * 16 + n;
        float bzv = (col < Hn) ? bz[col] : 0.0f;
        float bhv = (col < Hn) ? bh[col] : 0.0f;
#pragma unroll
        for (int i = 0; i < 8; ++i) {
            int row = row0 + i + 8 * hi;
            float z  = fast_sigmoid(accZ[t][i] + bzv);
            float hc = fast_tanh(accH[t][i] + bhv);
            float s  = (col < Hn) ? sbuf[(size_t)row * Hn + col] : 0.0f;
            float mn = (1.0f - z) * s + z * hc;
            int eid = lev_row[row];
            if (col < Hn && eid < En) mbuf[(size_t)eid * Hn + col] = mn;
            mnew16[(size_t)row * KP1 + col] = (col < Hn) ? (_Float16)mn : (_Float16)0.0f;
        }
    }
}

// r GEMMs + commit: rm = r*m_new, atomically accumulate node sums
__global__ __launch_bounds__(32)
void level_r_kernel(const _Float16* __restrict__ dx16, const _Float16* __restrict__ mnew16,
                    const _Float16* __restrict__ WrF, const _Float16* __restrict__ UrF,
                    const float* __restrict__ bUr, const int* __restrict__ lev_row,
                    const int* __restrict__ dstA,
                    const float* __restrict__ mbuf, float* __restrict__ rmbuf,
                    float* __restrict__ node_m, float* __restrict__ node_rm) {
    int ctg = blockIdx.x, rt = blockIdx.y;
    int lane = threadIdx.x & 31;
    int n = lane & 15, hi = lane >> 4;
    int row0 = rt * 16;
    v8f acc[NTW] = {};
    const _Float16* a1 = dx16   + (size_t)row0 * KP1;
    const _Float16* a2 = mnew16 + (size_t)row0 * KP1;
    for (int kt = 0; kt < KT1; ++kt) {
        v16h A1 = wmma_load_a(a1 + kt * 32, KP1);
        v16h A2 = wmma_load_a(a2 + kt * 32, KP1);
#pragma unroll
        for (int t = 0; t < NTW; ++t) {
            int ct = ctg * NTW + t;
            v16h Br = wmma_load_bfrag(WrF, kt, ct);
            acc[t] = __builtin_amdgcn_wmma_f32_16x16x32_f16(false, A1, false, Br, (short)0, acc[t], false, false);
            v16h Bu = wmma_load_bfrag(UrF, kt, ct);
            acc[t] = __builtin_amdgcn_wmma_f32_16x16x32_f16(false, A2, false, Bu, (short)0, acc[t], false, false);
        }
    }
#pragma unroll
    for (int t = 0; t < NTW; ++t) {
        int col = (ctg * NTW + t) * 16 + n;
        float bv = (col < Hn) ? bUr[col] : 0.0f;
#pragma unroll
        for (int i = 0; i < 8; ++i) {
            int row = row0 + i + 8 * hi;
            int eid = lev_row[row];
            if (col < Hn && eid < En) {
                float r  = fast_sigmoid(acc[t][i] + bv);
                float mn = mbuf[(size_t)eid * Hn + col];
                float rv = r * mn;
                rmbuf[(size_t)eid * Hn + col] = rv;
                int d = dstA[eid];
                atomicAdd(&node_m [(size_t)d * Hn + col], mn);
                atomicAdd(&node_rm[(size_t)d * Hn + col], rv);
            }
        }
    }
}

// Gather root rows: catg = [x[root] | node_m[root]] in f16
__global__ __launch_bounds__(256)
void stage_final_kernel(const int* __restrict__ roots, const float* __restrict__ xp,
                        const float* __restrict__ node_m, _Float16* __restrict__ catg) {
    int b = blockIdx.x;
    int node = roots[b];
    for (int c = threadIdx.x; c < KP2; c += blockDim.x) {
        float v = 0.0f;
        if (c < Hn)            v = xp[(size_t)node * Hn + c];
        else if (c < 2 * Hn)   v = node_m[(size_t)node * Hn + (c - Hn)];
        catg[(size_t)b * KP2 + c] = (_Float16)v;
    }
}

// out = relu(catg @ Wg + bg), rows = 2048 roots
__global__ __launch_bounds__(32)
void final_gemm_kernel(const _Float16* __restrict__ catg, const _Float16* __restrict__ WgF,
                       const float* __restrict__ bg, float* __restrict__ out) {
    int ctg = blockIdx.x, rt = blockIdx.y;
    int lane = threadIdx.x & 31;
    int n = lane & 15, hi = lane >> 4;
    int row0 = rt * 16;
    v8f acc[NTW] = {};
    const _Float16* a = catg + (size_t)row0 * KP2;
    for (int kt = 0; kt < KT2; ++kt) {
        v16h A = wmma_load_a(a + kt * 32, KP2);
#pragma unroll
        for (int t = 0; t < NTW; ++t) {
            v16h B = wmma_load_bfrag(WgF, kt, ctg * NTW + t);
            acc[t] = __builtin_amdgcn_wmma_f32_16x16x32_f16(false, A, false, B, (short)0, acc[t], false, false);
        }
    }
#pragma unroll
    for (int t = 0; t < NTW; ++t) {
        int col = (ctg * NTW + t) * 16 + n;
        if (col < Hn) {
            float bv = bg[col];
#pragma unroll
            for (int i = 0; i < 8; ++i) {
                int row = row0 + i + 8 * hi;
                float v = acc[t][i] + bv;
                out[(size_t)row * Hn + col] = v > 0.0f ? v : 0.0f;
            }
        }
    }
}

// ---------------- host launch ---------------------------------------------
extern "C" void kernel_launch(void* const* d_in, const int* in_sizes, int n_in,
                              void* d_out, int out_size, void* d_ws, size_t ws_size,
                              hipStream_t stream) {
    const int*   wid   = (const int*)  d_in[0];
    const int*   src   = (const int*)  d_in[1];
    const int*   dstA  = (const int*)  d_in[2];
    const int*   rev   = (const int*)  d_in[3];
    const int*   lev   = (const int*)  d_in[4];
    const int*   roots = (const int*)  d_in[5];
    const float* emb   = (const float*)d_in[6];
    const float* Wz    = (const float*)d_in[7];
    const float* bz    = (const float*)d_in[8];
    const float* Wr    = (const float*)d_in[9];
    const float* Ur    = (const float*)d_in[10];
    const float* bUr   = (const float*)d_in[11];
    const float* Wh    = (const float*)d_in[12];
    const float* bh    = (const float*)d_in[13];
    const float* Wg    = (const float*)d_in[14];
    const float* bg    = (const float*)d_in[15];

    // active edges per level (static from COUNTS): bottom-up d=5..1, top-down d=1..5
    static const int PL[NLEV] = {10240, 10240, 8192, 6144, 4096,
                                 4096, 6144, 8192, 10240, 10240};

    // bump-allocate workspace (256B aligned)
    char* ws = (char*)d_ws;
    auto alloc = [&](size_t bytes) -> char* {
        char* p = ws;
        ws += (bytes + 255) & ~(size_t)255;
        return p;
    };
    float*    xp      = (float*)   alloc((size_t)(Nn + 1) * Hn * 4);
    float*    mbuf    = (float*)   alloc((size_t)(En + 1) * Hn * 4);
    float*    rmbuf   = (float*)   alloc((size_t)(En + 1) * Hn * 4);
    float*    node_m  = (float*)   alloc((size_t)(Nn + 1) * Hn * 4);
    float*    node_rm = (float*)   alloc((size_t)(Nn + 1) * Hn * 4);
    float*    sbuf    = (float*)   alloc((size_t)PMAX * Hn * 4);
    _Float16* cat1    = (_Float16*)alloc((size_t)PMAX * KP2 * 2);
    _Float16* cat2    = (_Float16*)alloc((size_t)PMAX * KP2 * 2);
    _Float16* dx16    = (_Float16*)alloc((size_t)PMAX * KP1 * 2);
    _Float16* mnew16  = (_Float16*)alloc((size_t)PMAX * KP1 * 2);
    _Float16* WzF     = (_Float16*)alloc((size_t)KT2 * NCT * 512 * 2);
    _Float16* WhF     = (_Float16*)alloc((size_t)KT2 * NCT * 512 * 2);
    _Float16* WgF     = (_Float16*)alloc((size_t)KT2 * NCT * 512 * 2);
    _Float16* WrF     = (_Float16*)alloc((size_t)KT1 * NCT * 512 * 2);
    _Float16* UrF     = (_Float16*)alloc((size_t)KT1 * NCT * 512 * 2);
    _Float16* catg    = (_Float16*)alloc((size_t)Bn * KP2 * 2);

    // zero state each call (graph-capturable async memsets)
    hipMemsetAsync(mbuf,    0, (size_t)(En + 1) * Hn * 4, stream);
    hipMemsetAsync(rmbuf,   0, (size_t)(En + 1) * Hn * 4, stream);
    hipMemsetAsync(node_m,  0, (size_t)(Nn + 1) * Hn * 4, stream);
    hipMemsetAsync(node_rm, 0, (size_t)(Nn + 1) * Hn * 4, stream);
    hipMemsetAsync(xp + (size_t)Nn * Hn, 0, (size_t)Hn * 4, stream);   // dummy node row

    // x = emb[wid]
    embed_kernel<<<(int)(((size_t)Nn * Hn + 255) / 256), 256, 0, stream>>>(wid, emb, xp);

    // weights -> f16 fragment-major panels
    {
        int g2 = (int)(((size_t)KT2 * NCT * 512 + 255) / 256);
        int g1 = (int)(((size_t)KT1 * NCT * 512 + 255) / 256);
        convert_frag_kernel<<<g2, 256, 0, stream>>>(Wz, WzF, 2 * Hn, Hn, KT2);
        convert_frag_kernel<<<g2, 256, 0, stream>>>(Wh, WhF, 2 * Hn, Hn, KT2);
        convert_frag_kernel<<<g2, 256, 0, stream>>>(Wg, WgF, 2 * Hn, Hn, KT2);
        convert_frag_kernel<<<g1, 256, 0, stream>>>(Wr, WrF, Hn, Hn, KT1);
        convert_frag_kernel<<<g1, 256, 0, stream>>>(Ur, UrF, Hn, Hn, KT1);
    }

    for (int l = 0; l < NLEV; ++l) {
        const int* lrow = lev + (size_t)l * PMAX;
        int P = PL[l];
        stage_level_kernel<<<P, 256, 0, stream>>>(lrow, src, dstA, rev, xp,
                                                  node_m, node_rm, mbuf, rmbuf,
                                                  cat1, cat2, dx16, sbuf);
        level_zh_kernel<<<dim3(NCT / NTW, P / 16), 32, 0, stream>>>(
            cat1, cat2, WzF, WhF, bz, bh, sbuf, lrow, mbuf, mnew16);
        level_r_kernel<<<dim3(NCT / NTW, P / 16), 32, 0, stream>>>(
            dx16, mnew16, WrF, UrF, bUr, lrow, dstA, mbuf, rmbuf, node_m, node_rm);
    }

    stage_final_kernel<<<Bn, 256, 0, stream>>>(roots, xp, node_m, catg);
    final_gemm_kernel<<<dim3(NCT / NTW, Bn / 16), 32, 0, stream>>>(
        catg, WgF, bg, (float*)d_out);
}
